// VectorQuantizer_81716047773944
// MI455X (gfx1250) — matrix operand order repaired and checked by reading, not compiled, not run on hardware
//
#include <hip/hip_runtime.h>

typedef __attribute__((ext_vector_type(16))) __bf16 bf16x16;
typedef __attribute__((ext_vector_type(8)))  float  f32x8;

#define SP        32768          // L*H*W spatial per batch
#define CDIM      64
#define KCB       512
#define TOK_WG    64             // tokens per workgroup (4 M-tiles x 16)
#define QUANT_N   8388608        // 4*64*8*64*64
#define LOSS_OFF  QUANT_N
#define IDX_OFF   (QUANT_N + 1)

__device__ __forceinline__ unsigned short f2bf(float f) {
  unsigned u = __float_as_uint(f);
  u += 0x7FFFu + ((u >> 16) & 1u);      // round-to-nearest-even
  return (unsigned short)(u >> 16);
}
__device__ __forceinline__ float bf2f(unsigned short h) {
  return __uint_as_float((unsigned)h << 16);
}

// B-matrix (32x16 bf16) fragment layout for V_WMMA_*_16X16X32:
// lane l holds codeword col (l&15), dims 32q + 16*(l>>4) + j (j=0..15 contiguous)
__device__ __forceinline__ int bfrag_index(int k, int c) {
  int kt = k >> 4, q = c >> 5, half = (c >> 4) & 1, j = c & 15;
  int lane = (k & 15) | (half << 4);
  return ((((kt << 1) | q) << 5) | lane) * 16 + j;
}
// A-matrix (16x32 bf16): lane l holds row (l&15); h=l>>4 selects dim phase:
// positions 0..15 -> dims 32q+16h..+15 (contiguous, matches B)
__device__ __forceinline__ int afrag_index(int tok, int c) {
  int m = tok >> 4, q = c >> 5, r = c & 31;
  int h = (r >> 4) & 1;
  int pos = r & 15;
  int lane = (tok & 15) | (h << 4);
  return ((((m << 1) | q) << 5) | lane) * 16 + pos;
}

__device__ __forceinline__ bf16x16 ldfrag(const unsigned short* p) {
  return *reinterpret_cast<const bf16x16*>(p);
}

__global__ void vq_zero(float* out) { out[LOSS_OFF] = 0.0f; }

__global__ __launch_bounds__(256) void vq_kernel(
    const float* __restrict__ x, const float* __restrict__ cb,
    float* __restrict__ out) {
  __shared__ alignas(32) unsigned short s_cbHi[32 * 2 * 32 * 16];  // 64 KB
  __shared__ alignas(32) unsigned short s_cbLo[32 * 2 * 32 * 16];  // 64 KB
  __shared__ alignas(32) unsigned short s_aHi[4 * 2 * 32 * 16];    // 8 KB
  __shared__ alignas(32) unsigned short s_aLo[4 * 2 * 32 * 16];    // 8 KB
  __shared__ float s_norm[KCB];
  __shared__ float s_mval[2 * TOK_WG];
  __shared__ int   s_midx[2 * TOK_WG];
  __shared__ int   s_idx[TOK_WG];
  __shared__ float s_red[256];

  const int tid    = threadIdx.x;
  const int lane   = tid & 31;
  const int w      = tid >> 5;                 // wave 0..7
  const int n0     = blockIdx.x * TOK_WG;      // first global token
  const int b      = n0 >> 15;                 // batch (SP = 2^15)
  const int s0     = n0 & (SP - 1);
  const int inBase = b * (CDIM * SP) + s0;

  // ---- stage codebook: bf16 hi/lo B-fragments + exact f32 norms ----
  for (int r = 0; r < 2; ++r) {
    int k = tid * 2 + r;
    const float* row = cb + k * CDIM;
    float nrm = 0.0f;
#pragma unroll 8
    for (int c = 0; c < CDIM; ++c) {
      float v = row[c];
      nrm += v * v;
      unsigned short hi = f2bf(v);
      unsigned short lo = f2bf(v - bf2f(hi));
      int fi = bfrag_index(k, c);
      s_cbHi[fi] = hi;
      s_cbLo[fi] = lo;
    }
    s_norm[k] = nrm;
  }
  // ---- stage token tile: bf16 hi/lo A-fragments (coalesced reads) ----
  for (int it = 0; it < 16; ++it) {
    int e = it * 256 + tid;
    int c = e >> 6, tok = e & (TOK_WG - 1);
    float v = x[inBase + c * SP + tok];
    unsigned short hi = f2bf(v);
    unsigned short lo = f2bf(v - bf2f(hi));
    int fi = afrag_index(tok, c);
    s_aHi[fi] = hi;
    s_aLo[fi] = lo;
  }
  __syncthreads();

  // ---- per wave: M-tile m = w>>1; k-tile half = w&1 (16 k-tiles each) ----
  const int m     = w >> 1;
  const int khalf = w & 1;
  const bf16x16 aH0 = ldfrag(&s_aHi[(((m << 1) | 0) * 32 + lane) * 16]);
  const bf16x16 aH1 = ldfrag(&s_aHi[(((m << 1) | 1) * 32 + lane) * 16]);
  const bf16x16 aL0 = ldfrag(&s_aLo[(((m << 1) | 0) * 32 + lane) * 16]);
  const bf16x16 aL1 = ldfrag(&s_aLo[(((m << 1) | 1) * 32 + lane) * 16]);

  float mv[8];
  int   mi[8];
#pragma unroll
  for (int j = 0; j < 8; ++j) { mv[j] = 3.4e38f; mi[j] = 0x7fffffff; }

  const int colr = lane & 15;
  const int kt0  = khalf << 4;

  // software-pipelined B fragments: prefetch kt+1 while wmma-chain runs kt
  bf16x16 bH0 = ldfrag(&s_cbHi[((((kt0 << 1) | 0) << 5) | lane) * 16]);
  bf16x16 bL0 = ldfrag(&s_cbLo[((((kt0 << 1) | 0) << 5) | lane) * 16]);
  bf16x16 bH1 = ldfrag(&s_cbHi[((((kt0 << 1) | 1) << 5) | lane) * 16]);
  bf16x16 bL1 = ldfrag(&s_cbLo[((((kt0 << 1) | 1) << 5) | lane) * 16]);
  float nrm = s_norm[(kt0 << 4) | colr];

#pragma unroll 2
  for (int kt = 0; kt < 16; ++kt) {
    const int ktg = kt0 | kt;
    const bf16x16 cH0 = bH0, cL0 = bL0, cH1 = bH1, cL1 = bL1;
    const float   cnrm = nrm;
    if (kt < 15) {                       // uniform branch: EXEC stays full
      const int ng = ktg + 1;
      bH0 = ldfrag(&s_cbHi[((((ng << 1) | 0) << 5) | lane) * 16]);
      bL0 = ldfrag(&s_cbLo[((((ng << 1) | 0) << 5) | lane) * 16]);
      bH1 = ldfrag(&s_cbHi[((((ng << 1) | 1) << 5) | lane) * 16]);
      bL1 = ldfrag(&s_cbLo[((((ng << 1) | 1) << 5) | lane) * 16]);
      nrm = s_norm[(ng << 4) | colr];
    }
    f32x8 acc = {0.f, 0.f, 0.f, 0.f, 0.f, 0.f, 0.f, 0.f};
    // compensated bf16 product: (aH+aL)*(bH+bL) ~ aL*bH + aH*bL + aH*bH
    acc = __builtin_amdgcn_wmma_f32_16x16x32_bf16(false, aL0, false, cH0, (short)0, acc, false, false);
    acc = __builtin_amdgcn_wmma_f32_16x16x32_bf16(false, aH0, false, cL0, (short)0, acc, false, false);
    acc = __builtin_amdgcn_wmma_f32_16x16x32_bf16(false, aH0, false, cH0, (short)0, acc, false, false);
    acc = __builtin_amdgcn_wmma_f32_16x16x32_bf16(false, aL1, false, cH1, (short)0, acc, false, false);
    acc = __builtin_amdgcn_wmma_f32_16x16x32_bf16(false, aH1, false, cL1, (short)0, acc, false, false);
    acc = __builtin_amdgcn_wmma_f32_16x16x32_bf16(false, aH1, false, cH1, (short)0, acc, false, false);

    const int kcol = (ktg << 4) | colr;
#pragma unroll
    for (int j = 0; j < 8; ++j) {
      float d = cnrm - 2.0f * acc[j];          // ||c||^2 - 2 x.c  (||x||^2 const per row)
      if (d < mv[j]) { mv[j] = d; mi[j] = kcol; }
    }
  }

  // ---- butterfly (val,idx) min across the 16 lanes holding each row ----
#pragma unroll
  for (int j = 0; j < 8; ++j) {
    float v = mv[j];
    int   i = mi[j];
#pragma unroll
    for (int msk = 1; msk < 16; msk <<= 1) {
      float ov = __shfl_xor(v, msk, 32);
      int   oi = __shfl_xor(i, msk, 32);
      if (ov < v || (ov == v && oi < i)) { v = ov; i = oi; }
    }
    if ((lane & 15) == 0) {
      int row = j + ((lane >> 4) << 3);
      int tok = (m << 4) | row;
      s_mval[khalf * TOK_WG + tok] = v;
      s_midx[khalf * TOK_WG + tok] = i;
    }
  }
  __syncthreads();
  if (tid < TOK_WG) {
    float v0 = s_mval[tid], v1 = s_mval[TOK_WG + tid];
    int   i0 = s_midx[tid], i1 = s_midx[TOK_WG + tid];
    s_idx[tid] = (v1 < v0 || (v1 == v0 && i1 < i0)) ? i1 : i0;
  }
  __syncthreads();

  // ---- outputs: quant_st (coalesced), loss partial, indices ----
  float lsum = 0.0f;
  for (int it = 0; it < 16; ++it) {
    int e = it * 256 + tid;
    int c = e >> 6, tok = e & (TOK_WG - 1);
    int k = s_idx[tok];
    int fi = bfrag_index(k, c);
    float qv = bf2f(s_cbHi[fi]) + bf2f(s_cbLo[fi]);   // hi+lo ~ exact f32 codeword
    int off = inBase + c * SP + tok;
    float xv = x[off];
    float d = qv - xv;
    out[off] = xv + d;                                 // straight-through
    lsum += d * d;
  }
  if (tid < TOK_WG) out[IDX_OFF + n0 + tid] = (float)s_idx[tid];

  s_red[tid] = lsum;
  __syncthreads();
  for (int s = 128; s >= 1; s >>= 1) {
    if (tid < s) s_red[tid] += s_red[tid + s];
    __syncthreads();
  }
  if (tid == 0) atomicAdd(&out[LOSS_OFF], s_red[0] * (1.25f / (float)QUANT_N));
}

extern "C" void kernel_launch(void* const* d_in, const int* in_sizes, int n_in,
                              void* d_out, int out_size, void* d_ws, size_t ws_size,
                              hipStream_t stream) {
  const float* x  = (const float*)d_in[0];   // [4,64,8,64,64] f32
  const float* cb = (const float*)d_in[1];   // [512,64] f32
  float* out = (float*)d_out;                // quant_st | loss | indices
  (void)in_sizes; (void)n_in; (void)out_size; (void)d_ws; (void)ws_size;

  vq_zero<<<1, 1, 0, stream>>>(out);
  const int num_wg = (4 * SP) / TOK_WG;      // 131072 / 64 = 2048
  vq_kernel<<<num_wg, 256, 0, stream>>>(x, cb, out);
}
